// FMLayer_65524021068057
// MI455X (gfx1250) — matrix operand order repaired and checked by reading, not compiled, not run on hardware
//
#include <hip/hip_runtime.h>

typedef __attribute__((ext_vector_type(16))) __bf16 bf16x16;
typedef __attribute__((ext_vector_type(8)))  float  f32x8;
typedef __attribute__((ext_vector_type(4)))  int    i32x4;

#define NDIM 4096
#define KDIM 128
#define NCHUNK (NDIM / 32)   // 128 K-chunks of 32
#define NTILE  (KDIM / 16)   // 8 N-tiles of 16
#define CHUNK_ELEMS (NTILE * 512)   // 4096 bf16 = 8 KB per K-chunk

#if __has_builtin(__builtin_amdgcn_global_load_async_to_lds_b128)
#define FM_ASYNC 1
#else
#define FM_ASYNC 0
#endif

// ---------------------------------------------------------------------------
// Pre-pack V (NDIM x KDIM, f32 row-major) into bf16 WMMA B-fragments.
// Per (chunk c, tile nt): 32 lanes x 16 bf16 (32B/lane), B 32x16 in this code's contiguous k order:
// lanes 0-15 -> N=lane, K=0..15 (pairs per VGPR); lanes 16-31 -> K=16..31.
// ---------------------------------------------------------------------------
__global__ void fm_prepack(const float* __restrict__ V, __bf16* __restrict__ P) {
  int t    = blockIdx.x * blockDim.x + threadIdx.x;
  int lane = t & 31;
  int grp  = t >> 5;                       // grp = c*NTILE + nt
  if (grp >= NCHUNK * NTILE) return;
  int c  = grp >> 3;
  int nt = grp & 7;
  int n  = nt * 16 + (lane & 15);
  int kb = c * 32 + ((lane & 16) ? 16 : 0);
  __bf16* dst = P + (size_t)grp * 512 + (size_t)lane * 16;
#pragma unroll
  for (int j = 0; j < 16; ++j)
    dst[j] = (__bf16)V[(size_t)(kb + j) * KDIM + n];
}

// ---------------------------------------------------------------------------
// ss = || sum_k V[k,:] ||^2  (scalar)
// ---------------------------------------------------------------------------
__global__ void fm_colsumsq(const float* __restrict__ V, float* __restrict__ ss) {
  __shared__ float red[KDIM];
  int n = threadIdx.x;
  float s = 0.f;
  for (int k = 0; k < NDIM; ++k) s += V[(size_t)k * KDIM + n];
  red[n] = s * s;
  __syncthreads();
  for (int off = KDIM / 2; off > 0; off >>= 1) {
    if (n < off) red[n] += red[n + off];
    __syncthreads();
  }
  if (n == 0) ss[0] = red[0];
}

// ---------------------------------------------------------------------------
// Main FM kernel: 4 waves/CTA, 16 rows per wave, full K.
// B fragments double-buffered in LDS via async global->LDS copies.
// ---------------------------------------------------------------------------
__global__ __launch_bounds__(128)
void fm_main(const float* __restrict__ x, const float* __restrict__ W,
             const float* __restrict__ bptr, const __bf16* __restrict__ P,
             const float* __restrict__ ssptr, float* __restrict__ out) {
  __shared__ __bf16 Bb[2][CHUNK_ELEMS];          // 2 x 8 KB

  const int tid  = threadIdx.x;
  const int wave = tid >> 5;
  const int lane = tid & 31;
  const int half = lane >> 4;                    // 0: lanes 0-15, 1: 16-31
  const int ln   = lane & 15;
  const int rowBase = blockIdx.x * 64 + wave * 16;
  const float* xrow = x + (size_t)(rowBase + ln) * NDIM;

  // Issue the 8 KB copy of B-chunk c into LDS buffer `buf`.
  // 128 threads x 64B each (4 x async b128 per thread).
  auto issue_copy = [&](int c, int buf) {
    const __bf16* src = P + (size_t)c * CHUNK_ELEMS + (size_t)tid * 32;
    __bf16* dst = &Bb[buf][tid * 32];
#if FM_ASYNC
#pragma unroll
    for (int i = 0; i < 4; ++i)
      __builtin_amdgcn_global_load_async_to_lds_b128(
          (__attribute__((address_space(1))) i32x4*)(src + i * 8),
          (__attribute__((address_space(3))) i32x4*)(dst + i * 8), 0, 0);
#else
#pragma unroll
    for (int i = 0; i < 4; ++i)
      *(uint4*)(dst + i * 8) = *(const uint4*)(src + i * 8);
#endif
  };

  auto wait_async = [&]() {
#if FM_ASYNC
#if __has_builtin(__builtin_amdgcn_s_wait_asynccnt)
    __builtin_amdgcn_s_wait_asynccnt(0);
#else
    asm volatile("s_wait_asynccnt 0x0" ::: "memory");
#endif
#endif
  };

  f32x8 acc[NTILE] = {};               // 8 N-tiles of 16x16 f32 accumulators
  float rs = 0.f;                      // partial row sum
  float dw = 0.f;                      // partial x . W

  issue_copy(0, 0);                    // prologue: chunk 0 in flight

  for (int c = 0; c < NCHUNK; ++c) {
    // ---- A-side loads (start early; prefetch 8 chunks ahead) --------------
    const int k0 = c * 32 + half * 16; // A 16x32 contiguous: elems 0..7 -> K=16*half+0..7
    const int k1 = k0 + 8;             // elems 8..15 -> K=16*half+8..15
    __builtin_prefetch(xrow + k0 + 256, 0, 0);

    const float4 a0 = *(const float4*)(xrow + k0);
    const float4 a1 = *(const float4*)(xrow + k0 + 4);
    const float4 a2 = *(const float4*)(xrow + k1);
    const float4 a3 = *(const float4*)(xrow + k1 + 4);
    const float4 w0 = *(const float4*)(W + k0);
    const float4 w1 = *(const float4*)(W + k0 + 4);
    const float4 w2 = *(const float4*)(W + k1);
    const float4 w3 = *(const float4*)(W + k1 + 4);

    // ---- chunk c copy complete in all waves; buffers swap-safe ------------
    wait_async();
    __syncthreads();
    if (c + 1 < NCHUNK) issue_copy(c + 1, (c + 1) & 1);

    float av[16] = {a0.x, a0.y, a0.z, a0.w, a1.x, a1.y, a1.z, a1.w,
                    a2.x, a2.y, a2.z, a2.w, a3.x, a3.y, a3.z, a3.w};
    float wv[16] = {w0.x, w0.y, w0.z, w0.w, w1.x, w1.y, w1.z, w1.w,
                    w2.x, w2.y, w2.z, w2.w, w3.x, w3.y, w3.z, w3.w};

    bf16x16 af;
#pragma unroll
    for (int j = 0; j < 16; ++j) {
      af[j] = (__bf16)av[j];
      rs += av[j];
      dw = fmaf(av[j], wv[j], dw);
    }

    const __bf16* Bc = &Bb[c & 1][0];
#pragma unroll
    for (int nt = 0; nt < NTILE; ++nt) {
      bf16x16 bf = *(const bf16x16*)(Bc + nt * 512 + lane * 16);
      acc[nt] = __builtin_amdgcn_wmma_f32_16x16x32_bf16(
          false, af, false, bf, (short)0, acc[nt], false, false);
    }
  }

  // ---- per-row reductions --------------------------------------------------
  __shared__ float s_sq[4][16];
  __shared__ float s_rs[4][16];
  __shared__ float s_dw[4][16];
  if (lane < 16) {
    s_sq[wave][lane] = 0.f;
    s_rs[wave][lane] = 0.f;
    s_dw[wave][lane] = 0.f;
  }
  __syncthreads();

  // C layout: VGPR j, lanes 0-15 -> M=j ; lanes 16-31 -> M=j+8 ; N = ln+16*nt
#pragma unroll
  for (int j = 0; j < 8; ++j) {
    float p = 0.f;
#pragma unroll
    for (int nt = 0; nt < NTILE; ++nt) p += acc[nt][j] * acc[nt][j];
    atomicAdd(&s_sq[wave][half * 8 + j], p);   // ds_add_f32
  }
  atomicAdd(&s_rs[wave][ln], rs);
  atomicAdd(&s_dw[wave][ln], dw);
  __syncthreads();

  if (lane < 16) {
    const float b0 = bptr[0];
    const float ss = ssptr[0];
    const float r  = s_rs[wave][lane];
    out[rowBase + lane] =
        s_dw[wave][lane] + b0 + 0.5f * s_sq[wave][lane] - 0.5f * r * r * ss;
  }
}

// ---------------------------------------------------------------------------
extern "C" void kernel_launch(void* const* d_in, const int* in_sizes, int n_in,
                              void* d_out, int out_size, void* d_ws, size_t ws_size,
                              hipStream_t stream) {
  const float* x = (const float*)d_in[0];   // (16384, 4096)
  const float* W = (const float*)d_in[1];   // (1, 4096)
  const float* b = (const float*)d_in[2];   // (1,)
  const float* V = (const float*)d_in[3];   // (4096, 128)
  float* out = (float*)d_out;               // (16384, 1)

  // workspace layout: [0..255] ss scalar slot, [256..] packed bf16 V (1 MB)
  float*  ss = (float*)d_ws;
  __bf16* P  = (__bf16*)((char*)d_ws + 256);

  fm_prepack<<<(NCHUNK * NTILE * 32 + 255) / 256, 256, 0, stream>>>(V, P);
  fm_colsumsq<<<1, KDIM, 0, stream>>>(V, ss);
  fm_main<<<16384 / 64, 128, 0, stream>>>(x, W, b, P, ss, out);
}
